// ViT_82016695484522
// MI455X (gfx1250) — matrix operand order repaired and checked by reading, not compiled, not run on hardware
//
#include <hip/hip_runtime.h>
#include <stdint.h>

// ---------------- types ----------------
typedef __bf16 bf16_t;
typedef __attribute__((ext_vector_type(16))) __bf16 v16bf;
typedef __attribute__((ext_vector_type(8)))  __bf16 v8bf;
typedef __attribute__((ext_vector_type(8)))  float  v8f;
typedef unsigned int v4u __attribute__((ext_vector_type(4)));
typedef int          v4i __attribute__((ext_vector_type(4)));
typedef int          v8i __attribute__((ext_vector_type(8)));

// ---------------- model dims ----------------
enum : int {
  Bn = 32, CHn = 3, NPn = 14, PSn = 16,
  DIN = 768, HIDn = 768, HEADSn = 12, DHn = 64,
  NBLK = 12, FFn = 3072, OUTC = 1000,
  Sn = 197, PTn = 196, Spad = 208        // padded attn leading dim (16B rows)
};

// ---------------- TDM availability ----------------
#if defined(__has_builtin)
#if __has_builtin(__builtin_amdgcn_tensor_load_to_lds) && __has_builtin(__builtin_amdgcn_s_wait_tensorcnt)
#define HAVE_TDM 1
#endif
#endif
#ifndef HAVE_TDM
#define HAVE_TDM 0
#endif

#if HAVE_TDM
// One TDM descriptor: load a (rows x 32) bf16 tile (row stride = strideElems)
// into LDS at ldsAddr, 32 elems (64B) per row + 16B LDS pad per row
// (pad_interval code 3 = 16 DWORDs, pad_amount code 3 = 4 DWORDs -> 40-elem rows).
// OOB in K (remK) or rows (remRows) is zero-filled by the TDM.
__device__ __forceinline__ void tdm_load_tile(const bf16_t* gsrc, unsigned ldsAddr,
                                              int remK, int remRows, long long strideElems)
{
  const unsigned long long ga = (unsigned long long)(uintptr_t)gsrc;
  v4u g0;
  g0[0] = 1u;                                             // count=1, user mode
  g0[1] = ldsAddr;                                        // lds_addr
  g0[2] = (unsigned)(ga & 0xFFFFFFFFull);                 // global_addr[31:0]
  g0[3] = (unsigned)((ga >> 32) & 0x01FFFFFFull) | (2u << 30); // addr[56:32] | type=2
  const unsigned t0 = (unsigned)remK;                     // tensor_dim0 (K extent)
  const unsigned t1 = (unsigned)remRows;                  // tensor_dim1 (row extent)
  const unsigned long long st = (unsigned long long)strideElems;
  v8i g1;
  g1[0] = (int)((1u << 16) | (1u << 20) | (3u << 22) | (3u << 25)); // data_size=2B, pad_en, intvl=16DW, amt=4DW
  g1[1] = (int)((t0 & 0xFFFFu) << 16);                    // tensor_dim0[15:0]
  g1[2] = (int)((t0 >> 16) | ((t1 & 0xFFFFu) << 16));     // dim0[31:16] | dim1[15:0]
  g1[3] = (int)((t1 >> 16) | (32u << 16));                // dim1[31:16] | tile_dim0=32
  g1[4] = (int)128u;                                      // tile_dim1=128, tile_dim2=0
  g1[5] = (int)(st & 0xFFFFFFFFull);                      // dim0_stride[31:0]
  g1[6] = (int)((st >> 32) & 0xFFFFull);                  // dim0_stride[47:32]
  g1[7] = 0;
  v4i z4 = {0, 0, 0, 0};
#if defined(__clang_major__) && (__clang_major__ >= 23)
  v8i z8 = {0, 0, 0, 0, 0, 0, 0, 0};
  __builtin_amdgcn_tensor_load_to_lds(g0, g1, z4, z4, z8, 0);
#else
  __builtin_amdgcn_tensor_load_to_lds(g0, g1, z4, z4, 0);
#endif
}
#endif

// =====================================================================
//  Generic batched bf16 WMMA GEMM:  C = alpha * A(MxK) * Bt(NxK)^T + bias
//  Block tile 128(M) x 128(N), K-step 32, 256 threads (8 wave32 waves).
//  Each wave: 32x64 strip -> 2 A-frags x 4 B-frags = 8 WMMA per K-step.
//  Tiles staged to LDS by the Tensor Data Mover (one descriptor per tile).
//  Epilogue MODE:
//   0: f32 store            1: bf16 store
//   2: bf16 store + GELU    3: f32 accumulate (+=)
//   4: bf16 store transposed (for V): C[((m/S)*sC1 + z2*sC2 + n)*ldc + m%S]
// =====================================================================
template<int MODE>
__global__ __launch_bounds__(256)
void wmma_gemm(const bf16_t* __restrict__ A, int lda, long long sA1, long long sA2,
               const bf16_t* __restrict__ Bt, int ldb, long long sB1, long long sB2,
               void* __restrict__ Cv, int ldc, long long sC1, long long sC2,
               const float* __restrict__ bias, long long sb1, long long sb2,
               int M, int N, int K, int batch2, float alpha, int S_tok)
{
  const int zb = blockIdx.z / batch2;
  const int z2 = blockIdx.z % batch2;
  A  += (size_t)zb * sA1 + (size_t)z2 * sA2;
  Bt += (size_t)zb * sB1 + (size_t)z2 * sB2;
  if (bias) bias += (size_t)zb * sb1 + (size_t)z2 * sb2;

  const int m0 = blockIdx.y * 128;
  const int n0 = blockIdx.x * 128;

  __shared__ __align__(16) bf16_t As[128][40];   // 32 data + 8 pad elems per row
  __shared__ __align__(16) bf16_t Bs[128][40];

  const int tid   = threadIdx.x;
  const int wave  = tid >> 5;
  const int lane  = tid & 31;
  const int wm    = wave & 3;     // 4 waves along M (4 x 32 = 128)
  const int wn    = wave >> 2;    // 2 waves along N (2 x 64 = 128)
  const int lhalf = lane >> 4;
  const int l16   = lane & 15;

  v8f acc[2][4];
  for (int s = 0; s < 2; ++s)
    for (int f = 0; f < 4; ++f)
      for (int e = 0; e < 8; ++e) acc[s][f][e] = 0.0f;

#if HAVE_TDM
  const unsigned ldsA = (unsigned)(uintptr_t)&As[0][0];
  const unsigned ldsB = (unsigned)(uintptr_t)&Bs[0][0];
#endif

  for (int k0 = 0; k0 < K; k0 += 32) {
#if HAVE_TDM
    if (wave == 0) {
      tdm_load_tile(A  + (size_t)m0 * lda + k0, ldsA, K - k0, M - m0, lda);
      tdm_load_tile(Bt + (size_t)n0 * ldb + k0, ldsB, K - k0, N - n0, ldb);
      __builtin_amdgcn_s_wait_tensorcnt(0);
    }
#else
    // manual staging fallback: 128 rows x 4 chunks per tile, 2+2 chunks/thread
    for (int it = 0; it < 2; ++it) {
      const int idx = tid + it * 256;
      const int row = idx >> 2;
      const int cg  = (idx & 3) * 8;
      const int gm  = m0 + row;
      const bf16_t* src = A + (size_t)gm * lda + k0 + cg;
      bf16_t* dst = &As[row][cg];
      const bool inM  = gm < M;
      const bool full = inM && (k0 + cg + 8 <= K);
      if (full && ((((uintptr_t)src) & 15) == 0)) {
        *(v8bf*)dst = *(const v8bf*)src;
      } else {
        for (int e = 0; e < 8; ++e) {
          const int gk = k0 + cg + e;
          dst[e] = (inM && gk < K) ? src[e] : (bf16_t)0.0f;
        }
      }
    }
    for (int it = 0; it < 2; ++it) {
      const int idx = tid + it * 256;
      const int row = idx >> 2;
      const int cg  = (idx & 3) * 8;
      const int gn  = n0 + row;
      const bf16_t* src = Bt + (size_t)gn * ldb + k0 + cg;
      bf16_t* dst = &Bs[row][cg];
      const bool inN  = gn < N;
      const bool full = inN && (k0 + cg + 8 <= K);
      if (full && ((((uintptr_t)src) & 15) == 0)) {
        *(v8bf*)dst = *(const v8bf*)src;
      } else {
        for (int e = 0; e < 8; ++e) {
          const int gk = k0 + cg + e;
          dst[e] = (inN && gk < K) ? src[e] : (bf16_t)0.0f;
        }
      }
    }
#endif
    __syncthreads();

    // A fragments (16x32 bf16): VGPR0-3 K=khalf*8+0..7, VGPR4-7 K=16+khalf*8+0..7
    v16bf afr[2];
    for (int s = 0; s < 2; ++s) {
      const bf16_t* ap = &As[wm * 32 + s * 16 + l16][0];
      const v8bf alo = *(const v8bf*)(ap + lhalf * 8);
      const v8bf ahi = *(const v8bf*)(ap + 16 + lhalf * 8);
      afr[s] = __builtin_shufflevector(alo, ahi,
          0,1,2,3,4,5,6,7,8,9,10,11,12,13,14,15);
    }
    for (int f = 0; f < 4; ++f) {
      // B fragment (32x16 bf16): VGPR0-3 K=khalf*8+0..7, VGPR4-7 K=16+khalf*8+0..7
      const bf16_t* bp = &Bs[wn * 64 + f * 16 + l16][0];
      const v8bf blo = *(const v8bf*)(bp + lhalf * 8);
      const v8bf bhi = *(const v8bf*)(bp + 16 + lhalf * 8);
      const v16bf bfr = __builtin_shufflevector(blo, bhi,
          0,1,2,3,4,5,6,7,8,9,10,11,12,13,14,15);
      acc[0][f] = __builtin_amdgcn_wmma_f32_16x16x32_bf16(
          false, afr[0], false, bfr, (short)0, acc[0][f], false, false);
      acc[1][f] = __builtin_amdgcn_wmma_f32_16x16x32_bf16(
          false, afr[1], false, bfr, (short)0, acc[1][f], false, false);
    }
    __syncthreads();
  }

  // ---- epilogue ----
  const int gn_base = n0 + wn * 64;
  for (int s = 0; s < 2; ++s) {
    const int gm_base = m0 + wm * 32 + s * 16 + lhalf * 8;
    for (int f = 0; f < 4; ++f) {
      const int gn = gn_base + f * 16 + l16;
      if (gn >= N) continue;
      const float bvv = bias ? bias[gn] : 0.0f;
      for (int r = 0; r < 8; ++r) {
        const int gm = gm_base + r;
        if (gm >= M) continue;
        const float v = acc[s][f][r] * alpha + bvv;
        if constexpr (MODE == 0) {
          float* Cp = (float*)Cv + (size_t)zb * sC1 + (size_t)z2 * sC2;
          Cp[(size_t)gm * ldc + gn] = v;
        } else if constexpr (MODE == 1) {
          bf16_t* Cp = (bf16_t*)Cv + (size_t)zb * sC1 + (size_t)z2 * sC2;
          Cp[(size_t)gm * ldc + gn] = (bf16_t)v;
        } else if constexpr (MODE == 2) {
          const float g = 0.5f * v * (1.0f + erff(v * 0.70710678118654752f));
          bf16_t* Cp = (bf16_t*)Cv + (size_t)zb * sC1 + (size_t)z2 * sC2;
          Cp[(size_t)gm * ldc + gn] = (bf16_t)g;
        } else if constexpr (MODE == 3) {
          float* Cp = (float*)Cv + (size_t)zb * sC1 + (size_t)z2 * sC2;
          Cp[(size_t)gm * ldc + gn] += v;
        } else { // MODE == 4 : transposed V store
          bf16_t* Cp = (bf16_t*)Cv;
          const int b = gm / S_tok;
          const int ss = gm - b * S_tok;
          Cp[((size_t)b * sC1 + (size_t)z2 * sC2 + gn) * ldc + ss] = (bf16_t)v;
        }
      }
    }
  }
}

// =====================================================================
//  Elementwise / normalization kernels
// =====================================================================
__global__ __launch_bounds__(256)
void f32_to_bf16_kernel(const float* __restrict__ in, bf16_t* __restrict__ out, size_t n) {
  const size_t i = (size_t)blockIdx.x * 256 + threadIdx.x;
  if (i < n) out[i] = (bf16_t)in[i];
}

__global__ __launch_bounds__(256)
void patchify_kernel(const float* __restrict__ x, bf16_t* __restrict__ patches, int total) {
  const int i = blockIdx.x * 256 + threadIdx.x;
  if (i >= total) return;
  const int d   = i % DIN;
  const int rem = i / DIN;
  const int p   = rem % PTn;
  const int b   = rem / PTn;
  const int c   = d >> 8;
  const int iy  = (d & 255) >> 4;
  const int ix  = d & 15;
  const int py  = p / NPn, px = p % NPn;
  const int row = py * PSn + iy, col = px * PSn + ix;
  patches[i] = (bf16_t)x[(((size_t)b * CHn + c) * 224 + row) * 224 + col];
}

__global__ __launch_bounds__(256)
void assemble_kernel(const float* __restrict__ tokens, const float* __restrict__ cls,
                     float* __restrict__ out, int total) {
  const int i = blockIdx.x * 256 + threadIdx.x;
  if (i >= total) return;
  const int j   = i % HIDn;
  const int rem = i / HIDn;
  const int s   = rem % Sn;
  const int b   = rem / Sn;
  const float base = (s == 0) ? cls[j]
                              : tokens[((size_t)b * PTn + (s - 1)) * HIDn + j];
  const float expo = (float)(j & ~1) / (float)HIDn;
  const float ang  = (float)s * powf(10000.0f, -expo);
  const float pe   = (j & 1) ? cosf(ang) : sinf(ang);
  out[i] = base + pe;
}

__global__ __launch_bounds__(256)
void layernorm_bf16_kernel(const float* __restrict__ x, const float* __restrict__ g,
                           const float* __restrict__ bt, bf16_t* __restrict__ y, int D) {
  const int row = blockIdx.x;
  const float* xr = x + (size_t)row * D;
  __shared__ float red[256];
  const int tid = threadIdx.x;
  float s = 0.0f;
  for (int j = tid; j < D; j += 256) s += xr[j];
  red[tid] = s; __syncthreads();
  for (int o = 128; o > 0; o >>= 1) { if (tid < o) red[tid] += red[tid + o]; __syncthreads(); }
  const float mu = red[0] / (float)D;
  __syncthreads();
  float v = 0.0f;
  for (int j = tid; j < D; j += 256) { const float d = xr[j] - mu; v += d * d; }
  red[tid] = v; __syncthreads();
  for (int o = 128; o > 0; o >>= 1) { if (tid < o) red[tid] += red[tid + o]; __syncthreads(); }
  const float rstd = rsqrtf(red[0] / (float)D + 1e-5f);
  for (int j = tid; j < D; j += 256)
    y[(size_t)row * D + j] = (bf16_t)((xr[j] - mu) * rstd * g[j] + bt[j]);
}

// rows softmax with independent in/out leading dims; pads y[L..Lpad) with zeros
template<typename T>
__global__ __launch_bounds__(256)
void softmax_rows_kernel(const float* __restrict__ x, int xld,
                         T* __restrict__ y, int yld, int L, int Lpad) {
  const int row = blockIdx.x;
  const float* xr = x + (size_t)row * xld;
  T* yr = y + (size_t)row * yld;
  __shared__ float red[256];
  const int tid = threadIdx.x;
  float m = -3.0e38f;
  for (int j = tid; j < L; j += 256) m = fmaxf(m, xr[j]);
  red[tid] = m; __syncthreads();
  for (int o = 128; o > 0; o >>= 1) { if (tid < o) red[tid] = fmaxf(red[tid], red[tid + o]); __syncthreads(); }
  const float mx = red[0];
  __syncthreads();
  float s = 0.0f;
  for (int j = tid; j < L; j += 256) s += __expf(xr[j] - mx);
  red[tid] = s; __syncthreads();
  for (int o = 128; o > 0; o >>= 1) { if (tid < o) red[tid] += red[tid + o]; __syncthreads(); }
  const float inv = 1.0f / red[0];
  for (int j = tid; j < Lpad; j += 256)
    yr[j] = (j < L) ? (T)(__expf(xr[j] - mx) * inv) : (T)0.0f;
}

__global__ __launch_bounds__(256)
void extract_cls_kernel(const float* __restrict__ out, bf16_t* __restrict__ cls_act, int total) {
  const int i = blockIdx.x * 256 + threadIdx.x;
  if (i >= total) return;
  const int j = i % HIDn, b = i / HIDn;
  cls_act[i] = (bf16_t)out[(size_t)b * Sn * HIDn + j];
}

// =====================================================================
//  Host orchestration
// =====================================================================
template<int MODE>
static void launch_gemm(hipStream_t st,
                        const bf16_t* A, int lda, long long sA1, long long sA2,
                        const bf16_t* Bt, int ldb, long long sB1, long long sB2,
                        void* Cv, int ldc, long long sC1, long long sC2,
                        const float* bias, long long sb1, long long sb2,
                        int M, int N, int K, int zTotal, int batch2,
                        float alpha, int S_tok)
{
  dim3 grid((N + 127) / 128, (M + 127) / 128, zTotal);
  wmma_gemm<MODE><<<grid, dim3(256), 0, st>>>(A, lda, sA1, sA2, Bt, ldb, sB1, sB2,
                                              Cv, ldc, sC1, sC2, bias, sb1, sb2,
                                              M, N, K, batch2, alpha, S_tok);
}

extern "C" void kernel_launch(void* const* d_in, const int* in_sizes, int n_in,
                              void* d_out, int out_size, void* d_ws, size_t ws_size,
                              hipStream_t stream) {
  (void)in_sizes; (void)n_in; (void)out_size; (void)ws_size;

  const float* x   = (const float*)d_in[0];
  const float* Wp  = (const float*)d_in[1];
  const float* bp  = (const float*)d_in[2];
  const float* cls = (const float*)d_in[3];
  const float* Wq  = (const float*)d_in[4];
  const float* bq  = (const float*)d_in[5];
  const float* Wk  = (const float*)d_in[6];
  const float* bk  = (const float*)d_in[7];
  const float* Wv  = (const float*)d_in[8];
  const float* bv  = (const float*)d_in[9];
  const float* g1  = (const float*)d_in[10];
  const float* b1n = (const float*)d_in[11];
  const float* g2  = (const float*)d_in[12];
  const float* b2n = (const float*)d_in[13];
  const float* W1  = (const float*)d_in[14];
  const float* bm1 = (const float*)d_in[15];
  const float* W2  = (const float*)d_in[16];
  const float* bm2 = (const float*)d_in[17];
  const float* Wo  = (const float*)d_in[18];
  const float* bo  = (const float*)d_in[19];

  // ---- bump allocator on workspace ----
  char* ws = (char*)d_ws;
  size_t off = 0;
  auto alloc = [&](size_t bytes) -> void* {
    off = (off + 255) & ~(size_t)255;
    void* p = ws + off;
    off += bytes;
    return p;
  };

  const size_t nWp   = (size_t)HIDn * DIN;
  const size_t nWqkv = (size_t)NBLK * HEADSn * DHn * DHn;
  const size_t nW1   = (size_t)NBLK * FFn * HIDn;
  const size_t nWo   = (size_t)OUTC * HIDn;

  bf16_t* wp_b = (bf16_t*)alloc(nWp   * 2);
  bf16_t* wq_b = (bf16_t*)alloc(nWqkv * 2);
  bf16_t* wk_b = (bf16_t*)alloc(nWqkv * 2);
  bf16_t* wv_b = (bf16_t*)alloc(nWqkv * 2);
  bf16_t* w1_b = (bf16_t*)alloc(nW1   * 2);
  bf16_t* w2_b = (bf16_t*)alloc(nW1   * 2);
  bf16_t* wo_b = (bf16_t*)alloc(nWo   * 2);

  const size_t nPatch = (size_t)Bn * PTn * HIDn;          // 32*196*768
  const size_t nOut   = (size_t)Bn * Sn  * HIDn;          // 32*197*768
  const size_t nAttnP = (size_t)Bn * HEADSn * Sn * Spad;  // padded attn rows
  const size_t nVt    = (size_t)Bn * HIDn * Spad;
  const size_t nH1    = (size_t)Bn * Sn * FFn;

  bf16_t* patches = (bf16_t*)alloc(nPatch * 2);
  float*  tokens  = (float*) alloc(nPatch * 4);
  float*  outbuf  = (float*) alloc(nOut   * 4);
  bf16_t* xs      = (bf16_t*)alloc(nOut   * 2);
  bf16_t* qb      = (bf16_t*)alloc(nOut   * 2);
  bf16_t* kb      = (bf16_t*)alloc(nOut   * 2);
  bf16_t* vtb     = (bf16_t*)alloc(nVt    * 2);           // (B, HID, Spad) transposed V
  float*  scores  = (float*) alloc(nAttnP * 4);
  bf16_t* probs   = (bf16_t*)alloc(nAttnP * 2);
  bf16_t* h1      = (bf16_t*)alloc(nH1    * 2);
  bf16_t* cls_act = (bf16_t*)alloc((size_t)Bn * HIDn * 2);
  float*  logits  = (float*) alloc((size_t)Bn * OUTC * 4);

  auto cvt = [&](const float* src, bf16_t* dst, size_t n) {
    f32_to_bf16_kernel<<<(unsigned)((n + 255) / 256), 256, 0, stream>>>(src, dst, n);
  };
  cvt(Wp, wp_b, nWp);
  cvt(Wq, wq_b, nWqkv);
  cvt(Wk, wk_b, nWqkv);
  cvt(Wv, wv_b, nWqkv);
  cvt(W1, w1_b, nW1);
  cvt(W2, w2_b, nW1);
  cvt(Wo, wo_b, nWo);

  // ---- patch embedding ----
  {
    const int tot = (int)nPatch;
    patchify_kernel<<<(tot + 255) / 256, 256, 0, stream>>>(x, patches, tot);
  }
  launch_gemm<0>(stream, patches, DIN, 0, 0, wp_b, DIN, 0, 0,
                 tokens, HIDn, 0, 0, bp, 0, 0,
                 Bn * PTn, HIDn, DIN, 1, 1, 1.0f, Sn);
  {
    const int tot = (int)nOut;
    assemble_kernel<<<(tot + 255) / 256, 256, 0, stream>>>(tokens, cls, outbuf, tot);
  }

  const int Mtok = Bn * Sn;                            // 6304
  const long long SSp = (long long)Sn * Spad;          // per-(b,h) attn row block
  const long long HSSp = (long long)HEADSn * SSp;

  for (int blk = 0; blk < NBLK; ++blk) {
    const bf16_t* wq_l = wq_b + (size_t)blk * HEADSn * DHn * DHn;
    const bf16_t* wk_l = wk_b + (size_t)blk * HEADSn * DHn * DHn;
    const bf16_t* wv_l = wv_b + (size_t)blk * HEADSn * DHn * DHn;
    const float*  bq_l = bq + (size_t)blk * HIDn;
    const float*  bk_l = bk + (size_t)blk * HIDn;
    const float*  bv_l = bv + (size_t)blk * HIDn;

    // norm1 -> xs (bf16)
    layernorm_bf16_kernel<<<Mtok, 256, 0, stream>>>(outbuf, g1 + (size_t)blk * HIDn,
                                                    b1n + (size_t)blk * HIDn, xs, HIDn);

    // Q, K: per-head (6304 x 64) = xs[:, h*64:+64] @ W^T + b
    launch_gemm<1>(stream, xs, HIDn, 0, DHn, wq_l, DHn, 0, (long long)DHn * DHn,
                   qb, HIDn, 0, DHn, bq_l, 0, DHn,
                   Mtok, DHn, DHn, HEADSn, HEADSn, 1.0f, Sn);
    launch_gemm<1>(stream, xs, HIDn, 0, DHn, wk_l, DHn, 0, (long long)DHn * DHn,
                   kb, HIDn, 0, DHn, bk_l, 0, DHn,
                   Mtok, DHn, DHn, HEADSn, HEADSn, 1.0f, Sn);
    // V: same GEMM, stored transposed (B, h*64+e, s) with ld = Spad
    launch_gemm<4>(stream, xs, HIDn, 0, DHn, wv_l, DHn, 0, (long long)DHn * DHn,
                   vtb, Spad, (long long)HIDn, (long long)DHn, bv_l, 0, DHn,
                   Mtok, DHn, DHn, HEADSn, HEADSn, 1.0f, Sn);

    // scores(b,h,s,t) = Q @ K^T * 1/8  (ld = Spad)
    launch_gemm<0>(stream,
                   qb, HIDn, (long long)Sn * HIDn, DHn,
                   kb, HIDn, (long long)Sn * HIDn, DHn,
                   scores, Spad, HSSp, SSp, nullptr, 0, 0,
                   Sn, Sn, DHn, Bn * HEADSn, HEADSn, 0.125f, Sn);

    // softmax over t -> probs bf16 (pads [197,208) with zeros)
    softmax_rows_kernel<bf16_t><<<Bn * HEADSn * Sn, 256, 0, stream>>>(
        scores, Spad, probs, Spad, Sn, Spad);

    // out += probs @ V   (per (b,h): (197x197)@(197x64), ragged K zero-filled)
    launch_gemm<3>(stream,
                   probs, Spad, HSSp, SSp,
                   vtb, Spad, (long long)HIDn * Spad, (long long)DHn * Spad,
                   outbuf, HIDn, (long long)Sn * HIDn, DHn, nullptr, 0, 0,
                   Sn, DHn, Sn, Bn * HEADSn, HEADSn, 1.0f, Sn);

    // norm2 -> xs
    layernorm_bf16_kernel<<<Mtok, 256, 0, stream>>>(outbuf, g2 + (size_t)blk * HIDn,
                                                    b2n + (size_t)blk * HIDn, xs, HIDn);

    // MLP1: h1 = gelu(xs @ W1^T + bm1)  (6304 x 3072)
    launch_gemm<2>(stream, xs, HIDn, 0, 0,
                   w1_b + (size_t)blk * FFn * HIDn, HIDn, 0, 0,
                   h1, FFn, 0, 0, bm1 + (size_t)blk * FFn, 0, 0,
                   Mtok, FFn, HIDn, 1, 1, 1.0f, Sn);

    // MLP2: out += h1 @ W2^T + bm2
    launch_gemm<3>(stream, h1, FFn, 0, 0,
                   w2_b + (size_t)blk * HIDn * FFn, FFn, 0, 0,
                   outbuf, HIDn, 0, 0, bm2 + (size_t)blk * HIDn, 0, 0,
                   Mtok, HIDn, FFn, 1, 1, 1.0f, Sn);
  }

  // ---- head: logits = out[:,0] @ Wo^T + bo ; softmax -> d_out ----
  {
    const int tot = Bn * HIDn;
    extract_cls_kernel<<<(tot + 255) / 256, 256, 0, stream>>>(outbuf, cls_act, tot);
  }
  launch_gemm<0>(stream, cls_act, HIDn, 0, 0, wo_b, HIDn, 0, 0,
                 logits, OUTC, 0, 0, bo, 0, 0,
                 Bn, OUTC, HIDn, 1, 1, 1.0f, Sn);
  softmax_rows_kernel<float><<<Bn, 256, 0, stream>>>(logits, OUTC, (float*)d_out, OUTC, OUTC, OUTC);
}
